// WITRAN_2DPSGMU_Encoder_ConvLSTM_24034636988646
// MI455X (gfx1250) — matrix operand order repaired and checked by reading, not compiled, not run on hardware
//
#include <hip/hip_runtime.h>

// ---- CDNA5 WMMA vector types -------------------------------------------------
typedef __attribute__((ext_vector_type(16))) __bf16 v16bf;
typedef __attribute__((ext_vector_type(8)))  float  v8f;
typedef __attribute__((ext_vector_type(4)))  int    v4i;

// ---- Problem constants -------------------------------------------------------
#define BATCH     16
#define NB        384           // num*batch = 24*16
#define SL        47            // scan length
#define KTRUE     432           // 9 taps * 48 channels (K order: k = tap*48 + ic)
#define KPAD      448           // padded K (14 chunks of 32)
#define KCHUNKS   14
#define NOC       96            // conv output channels
#define HSTATE    393216        // 384 * 1024 floats per recurrent-state buffer
#define OUT_BASE1 36962304      // 384*47*2048  (output_all_slice floats)
#define OUT_BASE2 37355520      // + 16*24*1024 (hidden_row_all floats)
// LDS comb layout: [img][padded spatial 10x10][ic 48]  (ic contiguous!)
#define SPITCH    48            // elements per spatial site
#define IPITCH    4800          // 100 * 48 per image

// ---- helpers -----------------------------------------------------------------
static __device__ __forceinline__ unsigned short f2bf(float f) {
  unsigned u = __float_as_uint(f);
  u += 0x7FFFu + ((u >> 16) & 1u);          // round-to-nearest-even
  return (unsigned short)(u >> 16);
}
static __device__ __forceinline__ float rcp_fast(float x) {
  return __builtin_amdgcn_rcpf(x);          // single v_rcp_f32
}
static __device__ __forceinline__ float sigf(float x) {
  return rcp_fast(1.0f + __expf(-x));       // v_exp + v_add + v_rcp
}
static __device__ __forceinline__ float tanh_fast(float x) {
#if __has_builtin(__builtin_amdgcn_tanhf)
  return __builtin_amdgcn_tanhf(x);         // hardware v_tanh_f32
#else
  float e = __expf(-2.0f * x);
  return (1.0f - e) * rcp_fast(1.0f + e);
#endif
}

// ---- init: zero recurrent state for step 0 -----------------------------------
__global__ void zero_h(float* __restrict__ h0, float* __restrict__ c0) {
  int i = blockIdx.x * blockDim.x + threadIdx.x;
  int stride = gridDim.x * blockDim.x;
  for (int j = i; j < HSTATE; j += stride) { h0[j] = 0.0f; c0[j] = 0.0f; }
}

// ---- init: f32 weights -> bf16 rows [oc=96][KPAD=448], K order (tap, ic) -----
__global__ void swizzle_w(const float* __restrict__ w, unsigned short* __restrict__ wb) {
  int idx = blockIdx.x * blockDim.x + threadIdx.x;
  if (idx >= NOC * KPAD) return;
  int n = idx / KPAD;
  int k = idx - n * KPAD;
  float v = 0.0f;
  if (k < KTRUE) {
    int tap = k / 48;
    int ic  = k - tap * 48;
    v = w[n * KTRUE + ic * 9 + tap];        // original layout (oc, ic, ky, kx)
  }
  wb[idx] = f2bf(v);
}

// ---- one scan step: implicit-GEMM conv (WMMA bf16) + gates + state update ----
__global__ __launch_bounds__(256) void step_kernel(
    const float* __restrict__ x, const float* __restrict__ bias,
    const unsigned short* __restrict__ wb,
    const float* __restrict__ hrow_in, const float* __restrict__ hcol_in,
    float* __restrict__ hrow_out, float* __restrict__ hcol_out,
    float* __restrict__ out, int s)
{
  // zero-padded comb tensor, transposed: [2 img][10*10 spatial][48 ic] bf16
  __shared__ __align__(16) unsigned short comb[2 * IPITCH];

  const int tid  = threadIdx.x;
  const int lane = tid & 31;
  const int wv   = tid >> 5;                  // wave id 0..7
  const int hsel = (lane < 16) ? 0 : 1;       // which half of the wave
  const int ln   = lane & 15;
  const int m_wg = blockIdx.x * 128;          // this WG's M-tile base
  const int n0   = blockIdx.x * 2;            // first image handled by this WG

  if (tid == 0) __builtin_prefetch(wb, 0, 1);

  // ---- zero the padded comb buffer (borders become the SAME padding) --------
  {
    v4i z = {0, 0, 0, 0};
    v4i* cz = (v4i*)comb;                     // 1200 x 16B
    #pragma unroll
    for (int i = 0; i < 5; ++i) {
      int j = tid + i * 256;
      if (j < 1200) cz[j] = z;
    }
  }
  __syncthreads();

  // ---- fill interior: x diagonal gather + h_row + h_col (bf16, transposed) --
  #pragma unroll
  for (int img = 0; img < 2; ++img) {
    const int n  = n0 + img;
    const int r  = n >> 4, bb = n & 15;
    const int cidx = s - r;                   // diagonal wavefront column
    const float* xs = x + (((size_t)((bb * 24 + r) * 24)) << 10);
    const size_t hb = (size_t)n << 10;
    #pragma unroll
    for (int i = 0; i < 12; ++i) {
      const int e   = tid + i * 256;          // 0..3071 = (ic 0..47, pos 0..63)
      const int ic  = e >> 6;
      const int pos = e & 63;
      float v;
      if (e < 1024)       v = ((unsigned)cidx < 24u) ? xs[((size_t)cidx << 10) + e] : 0.0f;
      else if (e < 2048)  v = hrow_in[hb + (e - 1024)];
      else                v = hcol_in[hb + (e - 2048)];
      // padded spatial site (h+1, w+1), ic contiguous within a site
      const int site = ((pos >> 3) + 1) * 10 + (pos & 7) + 1;
      comb[img * IPITCH + site * SPITCH + ic] = f2bf(v);
    }
  }
  __syncthreads();

  // ---- per-lane bases -------------------------------------------------------
  const int row  = wv * 16 + ln;              // local M row 0..127
  const int nl   = row >> 6;                  // image within WG
  const int rpos = row & 63;
  // anchor: image + output spatial site; tap adds (ky*10+kx)*48, run adds ic
  const int lanebase = nl * IPITCH + ((rpos >> 3) * 10 + (rpos & 7)) * SPITCH
                     + hsel * 8;              // k0 offset maps to ic offset
  const unsigned short* bbase = wb + ln * KPAD + hsel * 8;

  v8f acc[6];
  #pragma unroll
  for (int j = 0; j < 6; ++j)
    #pragma unroll
    for (int e = 0; e < 8; ++e) acc[j][e] = 0.0f;

  // ---- fully unrolled K loop: A runs are single aligned ds_load_b128 --------
  #pragma unroll
  for (int kc = 0; kc < KCHUNKS; ++kc) {
    // k0 = kc*32 + hsel*8 (+16 for 2nd run); tap independent of hsel*8
    const int t0 = (kc * 32) / 48;
    const int t1 = (kc * 32 + 16) / 48;
    const int C0 = (kc * 32 - 48 * t0)      + ((t0 / 3) * 10 + (t0 % 3)) * SPITCH;
    const int C1 = (kc * 32 + 16 - 48 * t1) + ((t1 / 3) * 10 + (t1 % 3)) * SPITCH;

    union { v16bf v; v4i q[2]; } af;
    af.q[0] = *(const v4i*)(comb + lanebase + C0);       // 8 contiguous ic
    if (kc < KCHUNKS - 1) {
      af.q[1] = *(const v4i*)(comb + lanebase + C1);     // 8 contiguous ic
    } else {
      v4i z = {0, 0, 0, 0};
      af.q[1] = z;                                       // K 432..447 zero pad
    }

    #pragma unroll
    for (int j = 0; j < 6; ++j) {
      union { v16bf v; v4i q[2]; } bf_;
      const v4i* bp = (const v4i*)(bbase + j * (16 * KPAD) + kc * 32);
      bf_.q[0] = bp[0];                       // K 8h..8h+7, then 16+8h..16+8h+7
      bf_.q[1] = bp[2];
      acc[j] = __builtin_amdgcn_wmma_f32_16x16x32_bf16(
          false, af.v, false, bf_.v, (short)0, acc[j], false, false);
    }
  }

  // ---- epilogue: bias + gates + state update + all three outputs ------------
  const int c = ln;                            // channel within gate group
  const float b0 = bias[c],      b1 = bias[16 + c], b2 = bias[32 + c],
              b3 = bias[48 + c], b4 = bias[64 + c], b5 = bias[80 + c];
  #pragma unroll
  for (int v = 0; v < 8; ++v) {
    const int M   = hsel ? (v + 8) : v;
    const int m   = m_wg + wv * 16 + M;
    const int n   = m >> 6, pos = m & 63;
    const float ur  = sigf(acc[0][v] + b0);
    const float orr = sigf(acc[1][v] + b1);
    const float uc  = sigf(acc[2][v] + b2);
    const float ocg = sigf(acc[3][v] + b3);
    const float ir  = tanh_fast(acc[4][v] + b4);
    const float icv = tanh_fast(acc[5][v] + b5);
    const size_t hoff = ((size_t)n << 10) + c * 64 + pos;
    const float hrp = hrow_in[hoff];
    const float hcp = hcol_in[hoff];
    const float hrn = tanh_fast((1.0f - ur) * hrp + ur * ir) * orr;
    const float hcn = tanh_fast((1.0f - uc) * hcp + uc * icv) * ocg;
    hrow_out[hoff] = hrn;
    const int nr = (n + BATCH < NB) ? (n + BATCH) : (n + BATCH - NB);  // roll
    hcol_out[((size_t)nr << 10) + c * 64 + pos] = hcn;
    const size_t ob = (size_t)n * (SL * 2048) + (size_t)s * 2048 + c * 64 + pos;
    out[ob]        = hrn;          // output_all_slice row half
    out[ob + 1024] = hcn;          // output_all_slice col half
    if (s >= 23) {
      const int t = s - 23;
      const int r = n >> 4, bb = n & 15;
      if (r == t)
        out[(size_t)OUT_BASE1 + (size_t)bb * 24576 + t * 1024 + c * 64 + pos] = hrn;
      if (r == 23)
        out[(size_t)OUT_BASE2 + (size_t)bb * 24576 + t * 1024 + c * 64 + pos] = hcn;
    }
  }
}

// ---- host-side orchestration -------------------------------------------------
extern "C" void kernel_launch(void* const* d_in, const int* in_sizes, int n_in,
                              void* d_out, int out_size, void* d_ws, size_t ws_size,
                              hipStream_t stream) {
  (void)in_sizes; (void)n_in; (void)out_size; (void)ws_size;
  const float* x      = (const float*)d_in[0];
  const float* conv_w = (const float*)d_in[1];
  const float* conv_b = (const float*)d_in[2];
  float* out = (float*)d_out;

  float* ws    = (float*)d_ws;
  float* hrow0 = ws;
  float* hrow1 = ws + (size_t)HSTATE;
  float* hcol0 = ws + (size_t)2 * HSTATE;
  float* hcol1 = ws + (size_t)3 * HSTATE;
  unsigned short* wb = (unsigned short*)(ws + (size_t)4 * HSTATE);

  zero_h<<<256, 256, 0, stream>>>(hrow0, hcol0);
  swizzle_w<<<(NOC * KPAD + 255) / 256, 256, 0, stream>>>(conv_w, wb);

  for (int s = 0; s < SL; ++s) {
    float* hri = (s & 1) ? hrow1 : hrow0;
    float* hro = (s & 1) ? hrow0 : hrow1;
    float* hci = (s & 1) ? hcol1 : hcol0;
    float* hco = (s & 1) ? hcol0 : hcol1;
    step_kernel<<<192, 256, 0, stream>>>(x, conv_b, wb, hri, hci, hro, hco, out, s);
  }
}
